// ResidueGCN_61555471286487
// MI455X (gfx1250) — matrix operand order repaired and checked by reading, not compiled, not run on hardware
//
#include <hip/hip_runtime.h>
#include <hip/hip_bf16.h>

typedef __attribute__((ext_vector_type(16))) _Float16 v16h;
typedef __attribute__((ext_vector_type(8)))  _Float16 v8h;
typedef __attribute__((ext_vector_type(8)))  float    v8f;

union AFrag { v16h v; v8h h[2]; };

#define LDS_STRIDE 40   // f16 units: 80B row stride -> 16B aligned frags, conflict-free
#define A_BUF (128 * LDS_STRIDE)
#define B_BUF (64  * LDS_STRIDE)

// Accumulate acc += A[rowBase:+128, 0:K] @ B[0:K, colBase:+64].
// A: [M,K] f32 row-major, B: [K,Ncols] f32 row-major. f32->f16 on the fly, packed LDS
// stores, double-buffered LDS. Staging loads are issued AFTER the barrier so they are
// not drained by __syncthreads' workgroup fence and instead overlap the WMMA work.
__device__ __forceinline__ void gemm_accum(
    const float* __restrict__ A, const float* __restrict__ B,
    int M, int Ncols, int K, int rowBase, int colBase,
    _Float16* Alds, _Float16* Blds, v8f acc[2][2], int tid)
{
    const int lane = tid & 31;
    const int wave = tid >> 5;
    const int wm = (wave & 3) * 32;            // wave tile M offset (4 waves in M)
    const int wn = (wave >> 2) * 32;           // wave tile N offset (2 waves in N)
    const int ml = lane & 15;
    const int ka = (lane < 16) ? 0 : 8;        // A fragment K-half (ISA 16-bit A 16x32 layout)
    const int kb = (lane < 16) ? 0 : 8;        // B fragment K-half (same K order as A)

    // staging coordinates
    const int ac4 = tid & 7;                   // A: which float4 of the 32-wide K slice
    const int ar0 = tid >> 3;                  // A: base row (32 rows per pass, 4 passes)
    const int bn  = tid & 63;                  // B: column within tile
    const int bkp = (tid >> 6) * 2;            // B: even-k base (k-pairs, 4 passes of +8)

    const int nTiles = K >> 5;
    float4 fA[4];
    float  fB[8];

    auto stage = [&](int t) {                  // global -> registers for K-slice t
        const int k0 = t << 5;
        #pragma unroll
        for (int it = 0; it < 4; ++it) {
            int grow = rowBase + ar0 + it * 32;
            grow = (grow < M) ? grow : (M - 1);          // clamp tail rows (stores guarded)
            fA[it] = *(const float4*)(A + (size_t)grow * K + k0 + ac4 * 4);
        }
        #pragma unroll
        for (int it = 0; it < 4; ++it) {
            const int k = k0 + bkp + it * 8;
            fB[2 * it]     = B[(size_t)k * Ncols + colBase + bn];
            fB[2 * it + 1] = B[(size_t)(k + 1) * Ncols + colBase + bn];
        }
    };
    auto commit = [&](int buf) {               // registers -> LDS (packed f16 stores)
        _Float16* Ab = Alds + buf * A_BUF;
        _Float16* Bb = Blds + buf * B_BUF;
        #pragma unroll
        for (int it = 0; it < 4; ++it) {
            union { _Float16 h[4]; uint2 u; } p;
            p.h[0] = (_Float16)fA[it].x; p.h[1] = (_Float16)fA[it].y;
            p.h[2] = (_Float16)fA[it].z; p.h[3] = (_Float16)fA[it].w;
            *(uint2*)(Ab + (ar0 + it * 32) * LDS_STRIDE + ac4 * 4) = p.u;   // ds_store_b64
        }
        #pragma unroll
        for (int it = 0; it < 4; ++it) {
            union { _Float16 h[2]; unsigned u; } p;
            p.h[0] = (_Float16)fB[2 * it]; p.h[1] = (_Float16)fB[2 * it + 1];
            *(unsigned*)(Bb + bn * LDS_STRIDE + bkp + it * 8) = p.u;        // ds_store_b32
        }
    };

    stage(0);
    __syncthreads();           // previous users of the LDS buffers are done
    commit(0);

    int cur = 0;
    for (int t = 0; t < nTiles; ++t) {
        __syncthreads();                       // lds[cur] fully written, old readers done
        if (t + 1 < nTiles) stage(t + 1);      // issue loads AFTER the fence: fly under WMMA

        _Float16* Ab = Alds + cur * A_BUF;
        _Float16* Bb = Blds + cur * B_BUF;
        AFrag a[2], b[2];
        #pragma unroll
        for (int mt = 0; mt < 2; ++mt) {
            const _Float16* p = Ab + (wm + mt * 16 + ml) * LDS_STRIDE + ka;
            a[mt].h[0] = *(const v8h*)(p);          // K = ka..ka+7
            a[mt].h[1] = *(const v8h*)(p + 16);     // K = 16+ka..16+ka+7
        }
        #pragma unroll
        for (int nt = 0; nt < 2; ++nt) {
            const _Float16* p = Bb + (wn + nt * 16 + ml) * LDS_STRIDE + kb;
            b[nt].h[0] = *(const v8h*)(p);          // K = kb..kb+7
            b[nt].h[1] = *(const v8h*)(p + 16);     // K = 16+kb..16+kb+7
        }
        #pragma unroll
        for (int mt = 0; mt < 2; ++mt)
            #pragma unroll
            for (int nt = 0; nt < 2; ++nt)
                acc[mt][nt] = __builtin_amdgcn_wmma_f32_16x16x32_f16(
                    false, a[mt].v, false, b[nt].v, (short)0, acc[mt][nt], false, false);

        if (t + 1 < nTiles) commit(cur ^ 1);   // waits loadcnt here, after the WMMAs
        cur ^= 1;
    }
}

// C = act( A@B [+ A2@B2] + bias ), C: [M,Ncols] f32. Grid: (Ncols/64, ceil(M/128)), 256 thr.
__global__ __launch_bounds__(256) void sage_gemm(
    const float* __restrict__ A,  const float* __restrict__ B,
    const float* __restrict__ A2, const float* __restrict__ B2,
    const float* __restrict__ bias, float* __restrict__ C,
    int M, int Ncols, int K, int K2, int relu)
{
    __shared__ __align__(16) _Float16 Alds[2 * A_BUF];
    __shared__ __align__(16) _Float16 Blds[2 * B_BUF];
    const int tid = threadIdx.x;
    const int rowBase = blockIdx.y * 128;
    const int colBase = blockIdx.x * 64;

    v8f acc[2][2] = {};
    gemm_accum(A, B, M, Ncols, K, rowBase, colBase, Alds, Blds, acc, tid);
    if (A2)   // uniform branch: EXEC stays all-ones across WMMA
        gemm_accum(A2, B2, M, Ncols, K2, rowBase, colBase, Alds, Blds, acc, tid);

    // epilogue (ISA 32-bit C/D 16x16 layout: VGPR v -> M = v + (lane<16?0:8), N = lane%16)
    const int lane = tid & 31, wave = tid >> 5;
    const int wm = (wave & 3) * 32, wn = (wave >> 2) * 32;
    const int ml = lane & 15;
    const int rsel = (lane < 16) ? 0 : 8;
    #pragma unroll
    for (int mt = 0; mt < 2; ++mt)
        #pragma unroll
        for (int nt = 0; nt < 2; ++nt) {
            const int col = colBase + wn + nt * 16 + ml;
            const float bv = bias[col];
            #pragma unroll
            for (int v = 0; v < 8; ++v) {
                int row = rowBase + wm + mt * 16 + v + rsel;
                if (row < M) {
                    float x = acc[mt][nt][v] + bv;
                    if (relu) x = fmaxf(x, 0.0f);
                    C[(size_t)row * Ncols + col] = x;
                }
            }
        }
}

// Scatter-max over edges. Values are relu outputs (>=0), so unsigned-int atomicMax on the
// bit pattern == float max (maps to GLOBAL_ATOMIC_MAX_U32). Grid: E blocks, d/4 threads.
__global__ void edge_max(const float* __restrict__ m, const int* __restrict__ src,
                         const int* __restrict__ dst, float* __restrict__ neigh, int d)
{
    const int e = blockIdx.x;
    const int c = threadIdx.x * 4;
    const int s = src[e];
    const int t = dst[e];
    const float4 v = *(const float4*)(m + (size_t)s * d + c);
    unsigned* out = (unsigned*)(neigh + (size_t)t * d + c);
    atomicMax(out + 0, __float_as_uint(v.x));
    atomicMax(out + 1, __float_as_uint(v.y));
    atomicMax(out + 2, __float_as_uint(v.z));
    atomicMax(out + 3, __float_as_uint(v.w));
}

// Final layer: out[n] = sigmoid( x[n].Ws2 + neigh[n].Wn2 + b ). One wave per node.
__global__ __launch_bounds__(256) void sage_out(
    const float* __restrict__ x, const float* __restrict__ neigh,
    const float* __restrict__ ws, const float* __restrict__ wn,
    const float* __restrict__ b, float* __restrict__ out, int Nn, int d)
{
    const int wave = threadIdx.x >> 5, lane = threadIdx.x & 31;
    const int node = blockIdx.x * 8 + wave;
    if (node >= Nn) return;
    const float* xp = x + (size_t)node * d;
    const float* np = neigh + (size_t)node * d;
    float s = 0.0f;
    for (int k = lane; k < d; k += 32)
        s += xp[k] * ws[k] + np[k] * wn[k];
    #pragma unroll
    for (int off = 16; off > 0; off >>= 1)
        s += __shfl_xor(s, off, 32);
    if (lane == 0)
        out[node] = 1.0f / (1.0f + expf(-(s + b[0])));
}

extern "C" void kernel_launch(void* const* d_in, const int* in_sizes, int n_in,
                              void* d_out, int out_size, void* d_ws, size_t ws_size,
                              hipStream_t stream) {
    (void)in_sizes; (void)n_in; (void)out_size; (void)ws_size;
    const float* h   = (const float*)d_in[0];
    const int*   src = (const int*)d_in[1];
    const int*   dst = (const int*)d_in[2];
    const float* Wp0 = (const float*)d_in[3];
    const float* bp0 = (const float*)d_in[4];
    const float* Ws0 = (const float*)d_in[5];
    const float* Wn0 = (const float*)d_in[6];
    const float* b0  = (const float*)d_in[7];
    const float* Wp1 = (const float*)d_in[8];
    const float* bp1 = (const float*)d_in[9];
    const float* Ws1 = (const float*)d_in[10];
    const float* Wn1 = (const float*)d_in[11];
    const float* b1  = (const float*)d_in[12];
    const float* Wp2 = (const float*)d_in[13];
    const float* bp2 = (const float*)d_in[14];
    const float* Ws2 = (const float*)d_in[15];
    const float* Wn2 = (const float*)d_in[16];
    const float* b2  = (const float*)d_in[17];

    const int N = 10000, E = 160000;
    float* mbuf  = (float*)d_ws;                       // N*960
    float* neigh = mbuf  + (size_t)N * 960;            // N*960
    float* xA    = neigh + (size_t)N * 960;            // N*512
    float* xB    = xA    + (size_t)N * 512;            // N*512

    const dim3 blk(256);
    const int mblk = (N + 127) / 128;

    // ---- layer 0 (960 -> 512) ----
    sage_gemm<<<dim3(960 / 64, mblk), blk, 0, stream>>>(h, Wp0, nullptr, nullptr, bp0, mbuf,
                                                        N, 960, 960, 0, 1);
    hipMemsetAsync(neigh, 0, (size_t)N * 960 * sizeof(float), stream);
    edge_max<<<E, 960 / 4, 0, stream>>>(mbuf, src, dst, neigh, 960);
    sage_gemm<<<dim3(512 / 64, mblk), blk, 0, stream>>>(h, Ws0, neigh, Wn0, b0, xA,
                                                        N, 512, 960, 960, 1);

    // ---- layer 1 (512 -> 512) ----
    sage_gemm<<<dim3(512 / 64, mblk), blk, 0, stream>>>(xA, Wp1, nullptr, nullptr, bp1, mbuf,
                                                        N, 512, 512, 0, 1);
    hipMemsetAsync(neigh, 0, (size_t)N * 512 * sizeof(float), stream);
    edge_max<<<E, 512 / 4, 0, stream>>>(mbuf, src, dst, neigh, 512);
    sage_gemm<<<dim3(512 / 64, mblk), blk, 0, stream>>>(xA, Ws1, neigh, Wn1, b1, xB,
                                                        N, 512, 512, 512, 1);

    // ---- layer 2 (512 -> 1, sigmoid) ----
    sage_gemm<<<dim3(512 / 64, mblk), blk, 0, stream>>>(xB, Wp2, nullptr, nullptr, bp2, mbuf,
                                                        N, 512, 512, 0, 1);
    hipMemsetAsync(neigh, 0, (size_t)N * 512 * sizeof(float), stream);
    edge_max<<<E, 512 / 4, 0, stream>>>(mbuf, src, dst, neigh, 512);
    sage_out<<<(N + 7) / 8, blk, 0, stream>>>(xB, neigh, Ws2, Wn2, b2, (float*)d_out, N, 512);
}
